// SiamMot_54339926229563
// MI455X (gfx1250) — matrix operand order repaired and checked by reading, not compiled, not run on hardware
//
#include <hip/hip_runtime.h>

#define BB 2
#define HH 128
#define WW 128
#define CC 256
#define KK 64
#define RR 8
#define HW (HH*WW)

typedef __bf16 bf16_t;
typedef __attribute__((ext_vector_type(16))) __bf16        v16bf;
typedef __attribute__((ext_vector_type(8)))  float         v8f;
typedef __attribute__((ext_vector_type(4)))  unsigned int  u32x4;
typedef __attribute__((ext_vector_type(2)))  unsigned int  u32x2;

union Frag { v16bf v; u32x4 q[2]; };

__device__ __forceinline__ float silu_f(float x)    { return x / (1.f + __expf(-x)); }
__device__ __forceinline__ float sigmoid_f(float x) { return 1.f / (1.f + __expf(-x)); }

// ---------------------------------------------------------------------------
// zero per-batch max accumulator (must run every call: graph replay)
__global__ void k_zero_smax(unsigned int* smax) {
    if (threadIdx.x < BB) smax[threadIdx.x] = 0u;
}

// ---------------------------------------------------------------------------
// Pre-swizzle a (256,256,3,3) OIHW f32 weight into WMMA-B fragment order:
// wp[t][kc][nb][lane][e] = bf16( W[o = nb*16 + lane%16][i = kc*32 + (lane/16)*16 + e][t] )
// (B-matrix 32x16, contiguous k order matching A: lanes 0-15 hold K=0-15, lanes 16-31 hold K=16-31)
__global__ void k_pack_w(const float* __restrict__ w, bf16_t* __restrict__ wp) {
    int g = blockIdx.x * blockDim.x + threadIdx.x;
    if (g >= 9 * CC * CC) return;
    int e    =  g        & 15;
    int lane = (g >> 4)  & 31;
    int nb   = (g >> 9)  & 15;
    int kc   = (g >> 13) & 7;
    int t    =  g >> 16;
    int o = nb * 16 + (lane & 15);
    int i = kc * 32 + (lane >> 4) * 16 + e;
    wp[g] = (bf16_t)w[(o * CC + i) * 9 + t];
}

// NCHW f32 -> channels-last bf16
__global__ void k_cvt_l0(const float* __restrict__ l0, bf16_t* __restrict__ out) {
    int g = blockIdx.x * blockDim.x + threadIdx.x;
    if (g >= BB * HW * CC) return;
    int c  = g & 255;
    int bp = g >> 8;
    int p  = bp & (HW - 1);
    int b  = bp >> 14;
    out[g] = (bf16_t)l0[((size_t)(b * CC + c)) * HW + p];
}

// L2-normalize rows of length 256, emit bf16 (used for embeddings and tracklets)
__global__ void k_norm_rows(const float* __restrict__ in, bf16_t* __restrict__ out) {
    __shared__ float red[256];
    int row = blockIdx.x, c = threadIdx.x;
    float v = in[(size_t)row * CC + c];
    red[c] = v * v;
    __syncthreads();
    for (int s = 128; s > 0; s >>= 1) { if (c < s) red[c] += red[c + s]; __syncthreads(); }
    float r = 1.f / fmaxf(sqrtf(red[0]), 1e-12f);
    out[(size_t)row * CC + c] = (bf16_t)(v * r);
}

// ---------------------------------------------------------------------------
// correlation GEMM: y[b][k][hw] = relu( trk_n[b][k][:] . emb_n[b][hw][:] )
// block = 16-pixel tile x all 64 tracklets; 4 waves, one 16x16 m-tile each.
__global__ __launch_bounds__(128) void k_corr_wmma(const bf16_t* __restrict__ embn,
                                                   const bf16_t* __restrict__ trkn,
                                                   float* __restrict__ y) {
    int b    = blockIdx.x >> 10;
    int pix0 = (blockIdx.x & 1023) * 16;
    int lane = threadIdx.x & 31, wave = threadIdx.x >> 5;
    int hi = lane >> 4, m = lane & 15;
    int mbase = wave * 16;

    const bf16_t* arow = trkn + ((size_t)(b * KK + mbase + m)) * CC;
    const bf16_t* brow = embn + ((size_t)b * HW + pix0 + m) * CC;

    v8f acc = {};
#pragma unroll
    for (int kc = 0; kc < 8; ++kc) {
        Frag a, bf;
        a.q[0] = *(const u32x4*)(arow + kc * 32 + hi * 16);
        a.q[1] = *(const u32x4*)(arow + kc * 32 + hi * 16 + 8);
        bf.v   = *(const v16bf*)(brow + kc * 32 + hi * 16);
        acc = __builtin_amdgcn_wmma_f32_16x16x32_bf16(false, a.v, false, bf.v,
                                                      (short)0, acc, false, false);
    }
#pragma unroll
    for (int r = 0; r < 8; ++r) {
        int k = mbase + r + 8 * hi;
        y[((size_t)(b * KK + k)) * HW + pix0 + m] = fmaxf(acc[r], 0.f);
    }
}

// first-max argmax over HW per (b,k)
__global__ void k_argmax(const float* __restrict__ y, int* __restrict__ idx) {
    __shared__ float sv[256];
    __shared__ int   si[256];
    int tid = threadIdx.x;
    const float* row = y + (size_t)blockIdx.x * HW;
    float best = -1.f; int bi = 0;
    for (int i = tid; i < HW; i += 256) {
        float v = row[i];
        if (v > best) { best = v; bi = i; }
    }
    sv[tid] = best; si[tid] = bi;
    __syncthreads();
    for (int s = 128; s > 0; s >>= 1) {
        if (tid < s) {
            if (sv[tid + s] > sv[tid] ||
                (sv[tid + s] == sv[tid] && si[tid + s] < si[tid])) {
                sv[tid] = sv[tid + s]; si[tid] = si[tid + s];
            }
        }
        __syncthreads();
    }
    if (tid == 0) idx[blockIdx.x] = si[0];
}

// s[b][p] = sum_k y*mask; also atomic-max into smax[b] (non-negative float as uint)
__global__ void k_corr_sum(const float* __restrict__ y, const int* __restrict__ idx,
                           float* __restrict__ s, unsigned int* __restrict__ smax) {
    __shared__ int   ai[KK], aj[KK];
    __shared__ float red[256];
    int g = blockIdx.x * 256 + threadIdx.x;
    int b = g >> 14;
    int p = g & (HW - 1);
    if (threadIdx.x < KK) {
        int ix = idx[b * KK + threadIdx.x];
        ai[threadIdx.x] = ix / WW;
        aj[threadIdx.x] = ix % WW;
    }
    __syncthreads();
    int h = p >> 7, w = p & 127;
    float acc = 0.f;
    for (int k = 0; k < KK; ++k) {
        if (h >= ai[k] - RR && h < ai[k] + RR && w >= aj[k] - RR && w < aj[k] + RR)
            acc += y[((size_t)(b * KK + k)) * HW + p];
    }
    s[g] = acc;
    red[threadIdx.x] = acc;
    __syncthreads();
    for (int st = 128; st > 0; st >>= 1) {
        if (threadIdx.x < st) red[threadIdx.x] = fmaxf(red[threadIdx.x], red[threadIdx.x + st]);
        __syncthreads();
    }
    if (threadIdx.x == 0) atomicMax(smax + b, __float_as_uint(red[0]));
}

__global__ void k_corr_div(const float* __restrict__ s, const unsigned int* __restrict__ smax,
                           float* __restrict__ xF) {
    int g = blockIdx.x * 256 + threadIdx.x;
    int b = g >> 14;
    xF[g] = s[g] / fmaxf(__uint_as_float(smax[b]), 1e-30f);
}

// ---------------------------------------------------------------------------
// grouped ConvTranspose fuse terms as lhs-dilated gathers: out[b][p][o] = b2 + b3
__global__ void k_convt_fuse(const float* __restrict__ l1, const float* __restrict__ l2,
                             const float* __restrict__ w2, const float* __restrict__ w3,
                             float* __restrict__ outb) {
    int bp = blockIdx.x, o = threadIdx.x;
    int b = bp >> 14, p = bp & (HW - 1);
    int h = p >> 7, w = p & 127;
    float acc = 0.f;
    // p2: l1[B,512,64,64], lhs_dil=2, pad=2, k=4, groups=256 (2 in-ch/group)
    for (int ky = 0; ky < 4; ++ky) {
        int yy = h + ky - 2; if (yy < 0 || (yy & 1)) continue;
        int y2 = yy >> 1;    if (y2 >= 64) continue;
        for (int kx = 0; kx < 4; ++kx) {
            int xx = w + kx - 2; if (xx < 0 || (xx & 1)) continue;
            int x2 = xx >> 1;    if (x2 >= 64) continue;
#pragma unroll
            for (int i = 0; i < 2; ++i)
                acc += l1[(((size_t)(b * 2 * CC + o * 2 + i)) * 64 + y2) * 64 + x2] *
                       w2[((o * 2 + i) * 4 + ky) * 4 + kx];
        }
    }
    // p3: l2[B,1024,32,32], lhs_dil=4, pad=5, k=8, groups=256 (4 in-ch/group)
    for (int ky = 0; ky < 8; ++ky) {
        int yy = h + ky - 5; if (yy < 0 || (yy & 3)) continue;
        int y2 = yy >> 2;    if (y2 >= 32) continue;
        for (int kx = 0; kx < 8; ++kx) {
            int xx = w + kx - 5; if (xx < 0 || (xx & 3)) continue;
            int x2 = xx >> 2;    if (x2 >= 32) continue;
#pragma unroll
            for (int i = 0; i < 4; ++i)
                acc += l2[(((size_t)(b * 4 * CC + o * 4 + i)) * 32 + y2) * 32 + x2] *
                       w3[((o * 4 + i) * 8 + ky) * 8 + kx];
        }
    }
    outb[(size_t)bp * CC + o] = acc;
}

// ---------------------------------------------------------------------------
// 3x3 256->256 conv as implicit GEMM with WMMA bf16.
// Block: 4-row x 16-col pixel tile x all 256 out-channels (8 waves x 2 N-tiles,
// 4 row sub-tiles each). Each B-fragment pair is fetched once per (tap,kchunk)
// and reused across the 4 row sub-tiles -> 4x less L2 weight traffic.
// Epilogue: out = silu(acc + bias) [+ addb]; write bf16 and/or f32 (channels-last).
__global__ __launch_bounds__(256) void k_conv3x3_wmma(
        const bf16_t* __restrict__ in, const bf16_t* __restrict__ wp,
        const float* __restrict__ bias, const float* __restrict__ addb,
        bf16_t* __restrict__ obf, float* __restrict__ of32) {
    __shared__ bf16_t stage[6 * 18 * CC];   // rows row0-1 .. row0+4, cols w0-1 .. w0+16

    int blk  = blockIdx.x;                  // b * 256 + rg * 8 + ct
    int b    = blk >> 8;
    int rg   = (blk >> 3) & 31;
    int row0 = rg << 2;
    int w0   = (blk & 7) << 4;
    int tid  = threadIdx.x;

    // cooperative halo staging (zero-padded), 8-byte chunks
    for (int i = tid; i < 6 * 18 * (CC / 4); i += 256) {
        int r   = i / (18 * (CC / 4));
        int rem = i - r * 18 * (CC / 4);
        int p   = rem / (CC / 4);
        int cc4 = rem - p * (CC / 4);
        int gy = row0 - 1 + r;
        int gx = w0 - 1 + p;
        u32x2 val = {0u, 0u};
        if ((unsigned)gy < (unsigned)HH && (unsigned)gx < (unsigned)WW)
            val = *(const u32x2*)(in + (((size_t)(b * HH + gy)) * WW + gx) * CC + cc4 * 4);
        *(u32x2*)(stage + (r * 18 + p) * CC + cc4 * 4) = val;
    }
    __syncthreads();

    int lane = tid & 31, wave = tid >> 5;
    int hi = lane >> 4, m = lane & 15;
    int nb0 = wave * 2;
    v8f acc[8];                              // [row sub-tile][n-tile 0/1]
#pragma unroll
    for (int q = 0; q < 8; ++q) acc[q] = (v8f){};

    for (int t = 0; t < 9; ++t) {
        int dy = t / 3, dx = t - dy * 3;
        int colbase = (m + dx) * CC;
#pragma unroll
        for (int kc = 0; kc < 8; ++kc) {
            Frag b0, b1;
            const bf16_t* wb = wp + (((size_t)((t * 8 + kc) * 16 + nb0)) * 32 + lane) * 16;
            b0.v = *(const v16bf*)wb;
            b1.v = *(const v16bf*)(wb + 512);
#pragma unroll
            for (int r4 = 0; r4 < 4; ++r4) {
                const bf16_t* srow = stage + (r4 + dy) * (18 * CC) + colbase;
                Frag a;
                a.q[0] = *(const u32x4*)(srow + kc * 32 + hi * 16);
                a.q[1] = *(const u32x4*)(srow + kc * 32 + hi * 16 + 8);
                acc[r4 * 2]     = __builtin_amdgcn_wmma_f32_16x16x32_bf16(
                                      false, a.v, false, b0.v, (short)0, acc[r4 * 2], false, false);
                acc[r4 * 2 + 1] = __builtin_amdgcn_wmma_f32_16x16x32_bf16(
                                      false, a.v, false, b1.v, (short)0, acc[r4 * 2 + 1], false, false);
            }
        }
    }

#pragma unroll
    for (int r4 = 0; r4 < 4; ++r4) {
        size_t pixbase = ((size_t)(b * HH + row0 + r4)) * WW + w0;
#pragma unroll
        for (int r = 0; r < 8; ++r) {
            size_t pidx = (pixbase + r + 8 * hi) * CC;
#pragma unroll
            for (int nt = 0; nt < 2; ++nt) {
                int och = (nb0 + nt) * 16 + m;
                float v = silu_f(acc[r4 * 2 + nt][r] + bias[och]);
                if (addb) v += addb[pidx + och];
                if (of32) of32[pidx + och] = v;
                if (obf)  obf[pidx + och]  = (bf16_t)v;
            }
        }
    }
}

// ---------------------------------------------------------------------------
// h_c1a: 1->256 3x3 conv + SiLU. Block per pixel, thread per out channel.
__global__ void k_hc1a(const float* __restrict__ xF, const float* __restrict__ w,
                       const float* __restrict__ bsc, float* __restrict__ u) {
    __shared__ float nb9[9];
    int bp = blockIdx.x, o = threadIdx.x;
    int b = bp >> 14, p = bp & (HW - 1);
    int h = p >> 7, wq = p & 127;
    if (o < 9) {
        int y = h + o / 3 - 1, x = wq + o % 3 - 1;
        nb9[o] = ((unsigned)y < HH && (unsigned)x < WW) ? xF[b * HW + y * WW + x] : 0.f;
    }
    __syncthreads();
    float acc = bsc[o];
#pragma unroll
    for (int t = 0; t < 9; ++t) acc += nb9[t] * w[o * 9 + t];
    u[(size_t)bp * CC + o] = silu_f(acc);
}

// 256->1 3x3 conv on f32 channels-last input (h_c1b), no activation.
__global__ void k_conv_c1_f32(const float* __restrict__ in, const float* __restrict__ w,
                              const float* __restrict__ bsc, float* __restrict__ out) {
    __shared__ float red[256];
    int bp = blockIdx.x, o = threadIdx.x;
    int b = bp >> 14, p = bp & (HW - 1);
    int h = p >> 7, wq = p & 127;
    float acc = 0.f;
#pragma unroll
    for (int t = 0; t < 9; ++t) {
        int y = h + t / 3 - 1, x = wq + t % 3 - 1;
        if ((unsigned)y < HH && (unsigned)x < WW)
            acc += in[(((size_t)(b * HW)) + y * WW + x) * CC + o] * w[o * 9 + t];
    }
    red[o] = acc; __syncthreads();
    for (int s = 128; s > 0; s >>= 1) { if (o < s) red[o] += red[o + s]; __syncthreads(); }
    if (o == 0) out[bp] = red[0] + bsc[0];
}

// 256->1 3x3 conv on bf16 input + sigmoid (h_c2c -> hmmap output)
__global__ void k_conv_c1_bf_sig(const bf16_t* __restrict__ in, const float* __restrict__ w,
                                 const float* __restrict__ bsc, float* __restrict__ out) {
    __shared__ float red[256];
    int bp = blockIdx.x, o = threadIdx.x;
    int b = bp >> 14, p = bp & (HW - 1);
    int h = p >> 7, wq = p & 127;
    float acc = 0.f;
#pragma unroll
    for (int t = 0; t < 9; ++t) {
        int y = h + t / 3 - 1, x = wq + t % 3 - 1;
        if ((unsigned)y < HH && (unsigned)x < WW)
            acc += (float)in[(((size_t)(b * HW)) + y * WW + x) * CC + o] * w[o * 9 + t];
    }
    red[o] = acc; __syncthreads();
    for (int s = 128; s > 0; s >>= 1) { if (o < s) red[o] += red[o + s]; __syncthreads(); }
    if (o == 0) out[bp] = sigmoid_f(red[0] + bsc[0]);
}

// t = bf16( xf[b,p] * xd_hm[b,p,c] )
__global__ void k_mul(const float* __restrict__ xf, const float* __restrict__ xd,
                      bf16_t* __restrict__ t) {
    int g = blockIdx.x * 256 + threadIdx.x;
    if (g >= BB * HW * CC) return;
    t[g] = (bf16_t)(xf[g >> 8] * xd[g]);
}

// 256->4 3x3 conv (b_c2) -> siambox [B,H,W,4]
__global__ void k_c2_box(const bf16_t* __restrict__ in, const float* __restrict__ w,
                         const float* __restrict__ bsc, float* __restrict__ out) {
    __shared__ float red[256];
    int bp = blockIdx.x, o = threadIdx.x;
    int b = bp >> 14, p = bp & (HW - 1);
    int h = p >> 7, wq = p & 127;
    float aj[4] = {0.f, 0.f, 0.f, 0.f};
#pragma unroll
    for (int t = 0; t < 9; ++t) {
        int y = h + t / 3 - 1, x = wq + t % 3 - 1;
        if ((unsigned)y < HH && (unsigned)x < WW) {
            float v = (float)in[(((size_t)(b * HW)) + y * WW + x) * CC + o];
#pragma unroll
            for (int j = 0; j < 4; ++j) aj[j] += v * w[(j * CC + o) * 9 + t];
        }
    }
    for (int j = 0; j < 4; ++j) {
        red[o] = aj[j]; __syncthreads();
        for (int s = 128; s > 0; s >>= 1) { if (o < s) red[o] += red[o + s]; __syncthreads(); }
        if (o == 0) out[(size_t)bp * 4 + j] = red[0] + bsc[j];
        __syncthreads();
    }
}

// ---------------------------------------------------------------------------
extern "C" void kernel_launch(void* const* d_in, const int* in_sizes, int n_in,
                              void* d_out, int out_size, void* d_ws, size_t ws_size,
                              hipStream_t stream) {
    (void)in_sizes; (void)n_in; (void)out_size; (void)ws_size;
    const float* id_emb = (const float*)d_in[0];
    const float* trk    = (const float*)d_in[1];
    const float* l0     = (const float*)d_in[2];
    const float* l1     = (const float*)d_in[3];
    const float* l2     = (const float*)d_in[4];
    const float* b_p1w  = (const float*)d_in[5];  const float* b_p1b  = (const float*)d_in[6];
    const float* b_p2w  = (const float*)d_in[7];  const float* b_p3w  = (const float*)d_in[8];
    const float* b_c1w  = (const float*)d_in[9];  const float* b_c1b  = (const float*)d_in[10];
    const float* b_c2w  = (const float*)d_in[11]; const float* b_c2b  = (const float*)d_in[12];
    const float* h_p1w  = (const float*)d_in[13]; const float* h_p1b  = (const float*)d_in[14];
    const float* h_p2w  = (const float*)d_in[15]; const float* h_p3w  = (const float*)d_in[16];
    const float* h_c1aw = (const float*)d_in[17]; const float* h_c1ab = (const float*)d_in[18];
    const float* h_c1bw = (const float*)d_in[19]; const float* h_c1bb = (const float*)d_in[20];
    const float* h_c2aw = (const float*)d_in[21]; const float* h_c2ab = (const float*)d_in[22];
    const float* h_c2bw = (const float*)d_in[23]; const float* h_c2bb = (const float*)d_in[24];
    const float* h_c2cw = (const float*)d_in[25]; const float* h_c2cb = (const float*)d_in[26];

    char* ws = (char*)d_ws;
    size_t off = 0;
    auto alloc = [&](size_t bytes) -> char* {
        char* p = ws + off;
        off += (bytes + 255) & ~(size_t)255;
        return p;
    };

    const size_t PW = (size_t)9 * CC * CC * 2;
    bf16_t* wp_bp1 = (bf16_t*)alloc(PW);
    bf16_t* wp_bc1 = (bf16_t*)alloc(PW);
    bf16_t* wp_hp1 = (bf16_t*)alloc(PW);
    bf16_t* wp_h2a = (bf16_t*)alloc(PW);
    bf16_t* wp_h2b = (bf16_t*)alloc(PW);
    bf16_t* l0bf   = (bf16_t*)alloc((size_t)BB * HW * CC * 2);
    bf16_t* embn   = (bf16_t*)alloc((size_t)BB * HW * CC * 2);
    bf16_t* trkn   = (bf16_t*)alloc((size_t)BB * KK * CC * 2);
    float*  yb     = (float*) alloc((size_t)BB * KK * HW * 4);
    int*    idx    = (int*)   alloc((size_t)BB * KK * 4);
    float*  sbuf   = (float*) alloc((size_t)BB * HW * 4);
    unsigned* smax = (unsigned*)alloc(256);
    float*  xF     = (float*) alloc((size_t)BB * HW * 4);
    float*  xf2    = (float*) alloc((size_t)BB * HW * 4);
    float*  addbox = (float*) alloc((size_t)BB * HW * CC * 4);
    float*  addhm  = (float*) alloc((size_t)BB * HW * CC * 4);
    float*  xdhm   = (float*) alloc((size_t)BB * HW * CC * 4);
    bf16_t* xdbox  = (bf16_t*)alloc((size_t)BB * HW * CC * 2);
    bf16_t* xbbf   = (bf16_t*)alloc((size_t)BB * HW * CC * 2);
    // lifetime-based aliases (all safely dead before reuse)
    float*  ubuf = addbox;   // addbox consumed by b_p1 conv before hc1a writes
    bf16_t* tbf  = embn;     // embn consumed by corr GEMM before k_mul writes
    bf16_t* t2a  = l0bf;     // l0bf consumed by p1 convs before h_c2a writes
    bf16_t* t2b  = xdbox;    // xdbox consumed by b_c1 conv before h_c2b writes

    float* out_hm  = (float*)d_out;           // hmmap  [B,1,H,W]
    float* out_box = (float*)d_out + BB * HW; // siambox [B,H,W,4]

    k_zero_smax<<<1, 32, 0, stream>>>(smax);

    int pg = (9 * CC * CC + 255) / 256;
    k_pack_w<<<pg, 256, 0, stream>>>(b_p1w,  wp_bp1);
    k_pack_w<<<pg, 256, 0, stream>>>(b_c1w,  wp_bc1);
    k_pack_w<<<pg, 256, 0, stream>>>(h_p1w,  wp_hp1);
    k_pack_w<<<pg, 256, 0, stream>>>(h_c2aw, wp_h2a);
    k_pack_w<<<pg, 256, 0, stream>>>(h_c2bw, wp_h2b);

    k_cvt_l0<<<(BB * HW * CC) / 256, 256, 0, stream>>>(l0, l0bf);
    k_norm_rows<<<BB * HW, 256, 0, stream>>>(id_emb, embn);
    k_norm_rows<<<BB * KK, 256, 0, stream>>>(trk, trkn);

    k_corr_wmma<<<BB * (HW / 16), 128, 0, stream>>>(embn, trkn, yb);
    k_argmax<<<BB * KK, 256, 0, stream>>>(yb, idx);
    k_corr_sum<<<(BB * HW) / 256, 256, 0, stream>>>(yb, idx, sbuf, smax);
    k_corr_div<<<(BB * HW) / 256, 256, 0, stream>>>(sbuf, smax, xF);

    k_convt_fuse<<<BB * HW, 256, 0, stream>>>(l1, l2, b_p2w, b_p3w, addbox);
    k_convt_fuse<<<BB * HW, 256, 0, stream>>>(l1, l2, h_p2w, h_p3w, addhm);

    const int CONV_GRID = BB * (HH / 4) * 8;  // 512 blocks
    // box head
    k_conv3x3_wmma<<<CONV_GRID, 256, 0, stream>>>(l0bf,  wp_bp1, b_p1b, addbox, xdbox, nullptr);
    // heatmap fuse (needs f32 for broadcast multiply)
    k_conv3x3_wmma<<<CONV_GRID, 256, 0, stream>>>(l0bf,  wp_hp1, h_p1b, addhm,  nullptr, xdhm);
    // box conv1 + head
    k_conv3x3_wmma<<<CONV_GRID, 256, 0, stream>>>(xdbox, wp_bc1, b_c1b, nullptr, xbbf, nullptr);
    k_c2_box<<<BB * HW, 256, 0, stream>>>(xbbf, b_c2w, b_c2b, out_box);

    // heatmap head
    k_hc1a<<<BB * HW, 256, 0, stream>>>(xF, h_c1aw, h_c1ab, ubuf);
    k_conv_c1_f32<<<BB * HW, 256, 0, stream>>>(ubuf, h_c1bw, h_c1bb, xf2);
    k_mul<<<(BB * HW * CC) / 256, 256, 0, stream>>>(xf2, xdhm, tbf);
    k_conv3x3_wmma<<<CONV_GRID, 256, 0, stream>>>(tbf, wp_h2a, h_c2ab, nullptr, t2a, nullptr);
    k_conv3x3_wmma<<<CONV_GRID, 256, 0, stream>>>(t2a, wp_h2b, h_c2bb, nullptr, t2b, nullptr);
    k_conv_c1_bf_sig<<<BB * HW, 256, 0, stream>>>(t2b, h_c2cw, h_c2cb, out_hm);
}
